// FullLIVABLEPygModel_60292750901533
// MI455X (gfx1250) — matrix operand order repaired and checked by reading, not compiled, not run on hardware
//
#include <hip/hip_runtime.h>
#include <hip/hip_bf16.h>

// ---------------- CDNA5 WMMA types & helpers ----------------
typedef __attribute__((ext_vector_type(16))) __bf16        v16bf;
typedef __attribute__((ext_vector_type(8)))  float         v8f;
typedef __attribute__((ext_vector_type(4)))  unsigned int  uint4v;

union frag_u { uint4v q[2]; v16bf b; };

__device__ inline v8f wmma_bf16(v16bf a, v16bf b, v8f c) {
    // D = A(16x32 bf16) x B(32x16 bf16) + C(16x16 f32)
    return __builtin_amdgcn_wmma_f32_16x16x32_bf16(
        /*neg_a=*/false, a, /*neg_b=*/false, b,
        /*c_mod=*/(short)0, c, /*reuse_a=*/false, /*reuse_b=*/false);
}

// A-fragment: X row-major [M][K] bf16. Lane L: M=m0+(L&15), halfs i<8 -> K=k+8g+i,
// i>=8 -> K=k+8+8g+i  (matches ISA 16-bit A 16x32 layout). Two contiguous 16B loads.
__device__ inline v16bf load_afrag(const unsigned short* X, int ld, int m0, int k) {
    int lane = threadIdx.x & 31;
    int g = lane >> 4;
    const unsigned short* p = X + (size_t)(m0 + (lane & 15)) * ld + k + 8 * g;
    frag_u f;
    f.q[0] = *(const uint4v*)p;          // K = k+8g   .. k+8g+7
    f.q[1] = *(const uint4v*)(p + 16);   // K = k+8g+16.. k+8g+23
    return f.b;
}

// B-fragment for C = X*W^T: B[k][n] = W[n][k], W row-major [N][K] bf16.
// Lane L: N=n0+(L&15), halfs i<8 -> K=k+8g+i, i>=8 -> K=k+8+8g+i. Two contiguous 16B loads.
__device__ inline v16bf load_bfrag(const unsigned short* W, int ld, int n0, int k) {
    int lane = threadIdx.x & 31;
    int g = lane >> 4;
    const unsigned short* p = W + (size_t)(n0 + (lane & 15)) * ld + k + 8 * g;
    frag_u f;
    f.q[0] = *(const uint4v*)p;          // K = k+8g   .. k+8g+7
    f.q[1] = *(const uint4v*)(p + 16);   // K = k+8g+16.. k+8g+23
    return f.b;
}

__device__ inline unsigned short f2bf(float f) {   // round-to-nearest-even
    unsigned int u = __float_as_uint(f);
    unsigned int r = u + 0x7FFFu + ((u >> 16) & 1u);
    return (unsigned short)(r >> 16);
}

// ---------------- Problem constants ----------------
#define NN   20000
#define DD   768
#define EE   200000
#define BB   64
#define TT   128
#define SS   128
#define HH   512
#define CC   14
#define ALPHA_F 0.1f
#define OMA_F   0.9f

// ---------------- Elementwise utility kernels ----------------
__global__ void f2bf_k(const float* __restrict__ in, unsigned short* __restrict__ out, int n) {
    int i = blockIdx.x * blockDim.x + threadIdx.x;
    if (i < n) out[i] = f2bf(in[i]);
}

__global__ void fill_k(float* __restrict__ p, float v, int n) {
    int i = blockIdx.x * blockDim.x + threadIdx.x;
    if (i < n) p[i] = v;
}

// ---------------- GI = seq_bf @ Wih^T + b_ih  (both directions) ----------------
// M=8192 (B*T), N=1536 (3H), K=128. One wave -> 16x64 strip (4 N-tiles), K-loop of 4 WMMAs.
__global__ __launch_bounds__(256) void gi_gemm_k(
    const unsigned short* __restrict__ seqbf,
    const unsigned short* __restrict__ wihf, const unsigned short* __restrict__ wihb,
    const float* __restrict__ bihf, const float* __restrict__ bihb,
    float* __restrict__ gif, float* __restrict__ gib) {
    int job  = blockIdx.x * 8 + (threadIdx.x >> 5);   // 2 * 512 * 24 jobs
    int lane = threadIdx.x & 31;
    const int JPD = 512 * 24;
    int dir = job / JPD;
    int rem = job - dir * JPD;
    int mt = rem / 24, ng = rem % 24;
    const unsigned short* W = dir ? wihb : wihf;
    const float* bias       = dir ? bihb : bihf;
    float* GI               = dir ? gib  : gif;
    int m0 = mt * 16;
    v16bf a[4];
#pragma unroll
    for (int kk = 0; kk < 4; kk++) a[kk] = load_afrag(seqbf, SS, m0, kk * 32);
    int g = lane >> 4, nl = lane & 15;
#pragma unroll
    for (int j = 0; j < 4; j++) {
        int n0 = (ng * 4 + j) * 16;
        v8f acc = {};
#pragma unroll
        for (int kk = 0; kk < 4; kk++) {
            v16bf b = load_bfrag(W, SS, n0, kk * 32);
            acc = wmma_bf16(a[kk], b, acc);
        }
        float bv = bias[n0 + nl];
#pragma unroll
        for (int v = 0; v < 8; v++)
            GI[(size_t)(m0 + v + 8 * g) * 1536 + n0 + nl] = acc[v] + bv;
    }
}

// ---------------- One GRU timestep, both directions fused ----------------
// 8 blocks: blockIdx = dir*4 + mtile. GEMM gh = h_prev @ Whh^T via WMMA into LDS,
// then gates + running mean/max accumulation. Dynamic LDS = 16*1536*4 = 96KB (<320KB WGP).
__global__ __launch_bounds__(256) void gru_step_k(
    unsigned short* __restrict__ hbf, float* __restrict__ hf32,
    const unsigned short* __restrict__ whhf, const unsigned short* __restrict__ whhb,
    const float* __restrict__ gif, const float* __restrict__ gib,
    const float* __restrict__ bhhf, const float* __restrict__ bhhb,
    float* __restrict__ sum, float* __restrict__ mx, int t) {
    extern __shared__ float gh[];                     // [16][1536]
    int dir = blockIdx.x >> 2, mt = blockIdx.x & 3;
    int wave = threadIdx.x >> 5, lane = threadIdx.x & 31;
    const unsigned short* W = dir ? whhb : whhf;
    const unsigned short* H = hbf + dir * BB * HH;    // this dir's h_prev (bf16)
    int m0 = mt * 16;

    v8f acc[12];
#pragma unroll
    for (int j = 0; j < 12; j++) acc[j] = (v8f){};
    for (int kk = 0; kk < 16; kk++) {                 // K = 512 = 16 * 32
        v16bf a = load_afrag(H, HH, m0, kk * 32);
#pragma unroll
        for (int j = 0; j < 12; j++) {                // 96 N-tiles / 8 waves
            int nt = wave + 8 * j;
            v16bf b = load_bfrag(W, HH, nt * 16, kk * 32);
            acc[j] = wmma_bf16(a, b, acc[j]);
        }
    }
    int g = lane >> 4, nl = lane & 15;
#pragma unroll
    for (int j = 0; j < 12; j++) {
        int col = (wave + 8 * j) * 16 + nl;
#pragma unroll
        for (int v = 0; v < 8; v++) gh[(v + 8 * g) * 1536 + col] = acc[j][v];
    }
    __syncthreads();

    const float* GI  = dir ? gib  : gif;
    const float* bhh = dir ? bhhb : bhhf;
    int tt = dir ? (TT - 1 - t) : t;
    for (int i = threadIdx.x; i < 16 * HH; i += 256) {
        int r = i >> 9, c = i & (HH - 1);
        int M = m0 + r;                               // batch row 0..63
        size_t gir = ((size_t)M * TT + tt) * 1536;
        float i_r = GI[gir + c];
        float i_z = GI[gir + 512 + c];
        float i_n = GI[gir + 1024 + c];
        float h_r = gh[r * 1536 + c]        + bhh[c];
        float h_z = gh[r * 1536 + 512 + c]  + bhh[512 + c];
        float h_n = gh[r * 1536 + 1024 + c] + bhh[1024 + c];
        size_t hidx = (size_t)(dir * BB + M) * HH + c;
        float hp = hf32[hidx];
        float rg = 1.f / (1.f + expf(-(i_r + h_r)));
        float zg = 1.f / (1.f + expf(-(i_z + h_z)));
        float ng = tanhf(i_n + rg * h_n);
        float h2 = (1.f - zg) * ng + zg * hp;
        hf32[hidx] = h2;
        hbf[hidx]  = f2bf(h2);
        size_t sidx = (size_t)M * 1024 + dir * HH + c;
        sum[sidx] += h2;
        mx[sidx] = fmaxf(mx[sidx], h2);
    }
}

// ---------------- APPNP ----------------
__global__ void deg_k(const int* __restrict__ ei, float* __restrict__ deg, int E) {
    int i = blockIdx.x * blockDim.x + threadIdx.x;
    if (i < E) atomicAdd(&deg[ei[E + i]], 1.0f);      // dst degrees; self-loop pre-filled as 1.0
}

__global__ void dinv_k(const float* __restrict__ deg, float* __restrict__ dinv, int n) {
    int i = blockIdx.x * blockDim.x + threadIdx.x;
    if (i < n) dinv[i] = rsqrtf(fmaxf(deg[i], 1.0f));
}

// z_out = ALPHA*x + (1-ALPHA)*dinv^2*z_in   (teleport + self-loop term)
__global__ void hop_init_k(const float* __restrict__ x, const float* __restrict__ zi,
                           const float* __restrict__ dinv, float* __restrict__ zo, int total) {
    for (int i = blockIdx.x * blockDim.x + threadIdx.x; i < total; i += gridDim.x * blockDim.x) {
        int node = i / DD;
        float di = dinv[node];
        zo[i] = ALPHA_F * x[i] + OMA_F * di * di * zi[i];
    }
}

// scatter-add edge messages: zo[dst] += (1-ALPHA)*dinv[s]*dinv[d]*zi[src]
__global__ void hop_edge_k(const int* __restrict__ ei, const float* __restrict__ dinv,
                           const float* __restrict__ zi, float* __restrict__ zo, int E) {
    int total = E * (DD / 4);
    for (int i = blockIdx.x * blockDim.x + threadIdx.x; i < total; i += gridDim.x * blockDim.x) {
        int e = i / (DD / 4);
        int c = (i - e * (DD / 4)) * 4;
        int s = ei[e], d = ei[E + e];
        float w = OMA_F * dinv[s] * dinv[d];
        const float4 zs = *(const float4*)(zi + (size_t)s * DD + c);
        float* zd = zo + (size_t)d * DD + c;
        atomicAdd(zd + 0, w * zs.x);
        atomicAdd(zd + 1, w * zs.y);
        atomicAdd(zd + 2, w * zs.z);
        atomicAdd(zd + 3, w * zs.w);
    }
}

// ---------------- Pooling ----------------
__global__ void cnt_k(const int* __restrict__ batch, float* __restrict__ cnt, int n) {
    int i = blockIdx.x * blockDim.x + threadIdx.x;
    if (i < n) atomicAdd(&cnt[batch[i]], 1.0f);
}

__global__ void pool_k(const int* __restrict__ batch, const float* __restrict__ z,
                       float* __restrict__ pooled, int total) {
    for (int i = blockIdx.x * blockDim.x + threadIdx.x; i < total; i += gridDim.x * blockDim.x) {
        int node = i / DD;
        int c = i - node * DD;
        atomicAdd(&pooled[(size_t)batch[node] * DD + c], z[i]);
    }
}

__global__ void pooldiv_k(float* __restrict__ pooled, const float* __restrict__ cnt, int total) {
    int i = blockIdx.x * blockDim.x + threadIdx.x;
    if (i < total) pooled[i] /= fmaxf(cnt[i / DD], 1.0f);
}

// ---------------- Small fp32 linear layers (M=64) ----------------
__global__ void linear_k(const float* __restrict__ in, const float* __restrict__ W,
                         const float* __restrict__ b, float* __restrict__ out,
                         int M, int N, int K, int relu) {
    int idx = blockIdx.x * blockDim.x + threadIdx.x;
    if (idx >= M * N) return;
    int m = idx / N, n = idx - m * N;
    const float* ip = in + (size_t)m * K;
    const float* wp = W + (size_t)n * K;
    float acc = 0.f;
    for (int k = 0; k < K; k += 4) {
        float4 a = *(const float4*)(ip + k);
        float4 w = *(const float4*)(wp + k);
        acc = fmaf(a.x, w.x, fmaf(a.y, w.y, fmaf(a.z, w.z, fmaf(a.w, w.w, acc))));
    }
    acc += b[n];
    out[idx] = relu ? fmaxf(acc, 0.f) : acc;
}

__global__ void seqfeat_k(const float* __restrict__ sum, const float* __restrict__ mx,
                          float* __restrict__ feat, int total) {
    int i = blockIdx.x * blockDim.x + threadIdx.x;
    if (i < total) feat[i] = sum[i] * (1.0f / (float)TT) + mx[i];
}

__global__ void add_k(const float* __restrict__ a, const float* __restrict__ b,
                      float* __restrict__ o, int n) {
    int i = blockIdx.x * blockDim.x + threadIdx.x;
    if (i < n) o[i] = a[i] + b[i];
}

// ---------------- Host orchestration ----------------
static inline int LB(long n) { return (int)((n + 255) / 256); }

extern "C" void kernel_launch(void* const* d_in, const int* in_sizes, int n_in,
                              void* d_out, int out_size, void* d_ws, size_t ws_size,
                              hipStream_t stream) {
    (void)in_sizes; (void)n_in; (void)out_size; (void)ws_size;

    const float* x     = (const float*)d_in[0];
    const int*   ei    = (const int*)d_in[1];     // [2, E] (src row then dst row)
    const float* seq   = (const float*)d_in[2];
    const int*   batch = (const int*)d_in[3];
    const float* wih_f = (const float*)d_in[4];
    const float* whh_f = (const float*)d_in[5];
    const float* bih_f = (const float*)d_in[6];
    const float* bhh_f = (const float*)d_in[7];
    const float* wih_b = (const float*)d_in[8];
    const float* whh_b = (const float*)d_in[9];
    const float* bih_b = (const float*)d_in[10];
    const float* bhh_b = (const float*)d_in[11];
    const float* mg_w0 = (const float*)d_in[12];
    const float* mg_b0 = (const float*)d_in[13];
    const float* mg_w1 = (const float*)d_in[14];
    const float* mg_b1 = (const float*)d_in[15];
    const float* mg_w2 = (const float*)d_in[16];
    const float* mg_b2 = (const float*)d_in[17];
    const float* ms_w0 = (const float*)d_in[18];
    const float* ms_b0 = (const float*)d_in[19];
    const float* ms_w1 = (const float*)d_in[20];
    const float* ms_b1 = (const float*)d_in[21];
    const float* ms_w2 = (const float*)d_in[22];
    const float* ms_b2 = (const float*)d_in[23];
    float* out = (float*)d_out;

    // workspace carve-out (256B aligned)
    char* base = (char*)d_ws;
    size_t off = 0;
    auto take = [&](size_t bytes) -> char* {
        char* p = base + off;
        off += (bytes + 255) & ~(size_t)255;
        return p;
    };
    unsigned short* SEQBF = (unsigned short*)take((size_t)BB * TT * SS * 2);
    unsigned short* WIHF  = (unsigned short*)take((size_t)3 * HH * SS * 2);
    unsigned short* WIHB  = (unsigned short*)take((size_t)3 * HH * SS * 2);
    unsigned short* WHHF  = (unsigned short*)take((size_t)3 * HH * HH * 2);
    unsigned short* WHHB  = (unsigned short*)take((size_t)3 * HH * HH * 2);
    float* GIF   = (float*)take((size_t)BB * TT * 3 * HH * 4);   // 50.3 MB
    float* GIB   = (float*)take((size_t)BB * TT * 3 * HH * 4);   // 50.3 MB
    float* HF32  = (float*)take((size_t)2 * BB * HH * 4);
    unsigned short* HBF = (unsigned short*)take((size_t)2 * BB * HH * 2);
    float* SUMB  = (float*)take((size_t)BB * 1024 * 4);
    float* MAXB  = (float*)take((size_t)BB * 1024 * 4);
    float* DEG   = (float*)take((size_t)NN * 4);
    float* DINV  = (float*)take((size_t)NN * 4);
    float* Z0    = (float*)take((size_t)NN * DD * 4);            // 61.4 MB
    float* Z1    = (float*)take((size_t)NN * DD * 4);            // 61.4 MB
    float* CNT   = (float*)take((size_t)BB * 4);
    float* POOL  = (float*)take((size_t)BB * DD * 4);
    float* GH1   = (float*)take((size_t)BB * 384 * 4);
    float* GH2   = (float*)take((size_t)BB * 192 * 4);
    float* GLOG  = (float*)take((size_t)BB * CC * 4);
    float* SFEA  = (float*)take((size_t)BB * 1024 * 4);
    float* SH1   = (float*)take((size_t)BB * 512 * 4);
    float* SH2   = (float*)take((size_t)BB * 256 * 4);
    float* SLOG  = (float*)take((size_t)BB * CC * 4);

    // 1) bf16 conversions (sequence + GRU weights)
    f2bf_k<<<LB((long)BB * TT * SS), 256, 0, stream>>>(seq, SEQBF, BB * TT * SS);
    f2bf_k<<<LB(3 * HH * SS), 256, 0, stream>>>(wih_f, WIHF, 3 * HH * SS);
    f2bf_k<<<LB(3 * HH * SS), 256, 0, stream>>>(wih_b, WIHB, 3 * HH * SS);
    f2bf_k<<<LB(3 * HH * HH), 256, 0, stream>>>(whh_f, WHHF, 3 * HH * HH);
    f2bf_k<<<LB(3 * HH * HH), 256, 0, stream>>>(whh_b, WHHB, 3 * HH * HH);

    // 2) state init (must happen every call — graph replays don't re-poison)
    fill_k<<<LB(2 * BB * HH), 256, 0, stream>>>(HF32, 0.0f, 2 * BB * HH);
    fill_k<<<LB(BB * HH), 256, 0, stream>>>((float*)HBF, 0.0f, BB * HH); // 2*BB*HH bf16 = BB*HH f32
    fill_k<<<LB(BB * 1024), 256, 0, stream>>>(SUMB, 0.0f, BB * 1024);
    fill_k<<<LB(BB * 1024), 256, 0, stream>>>(MAXB, -3.0e38f, BB * 1024);
    fill_k<<<LB(NN), 256, 0, stream>>>(DEG, 1.0f, NN);                   // self-loop
    fill_k<<<LB(BB), 256, 0, stream>>>(CNT, 0.0f, BB);
    fill_k<<<LB(BB * DD), 256, 0, stream>>>(POOL, 0.0f, BB * DD);

    // 3) GI = seq @ Wih^T + b_ih for both directions (WMMA)
    gi_gemm_k<<<3072, 256, 0, stream>>>(SEQBF, WIHF, WIHB, bih_f, bih_b, GIF, GIB);

    // 4) sequential BiGRU: 128 fused steps (fwd t, bwd T-1-t), WMMA + gates + pooling
    for (int t = 0; t < TT; t++)
        gru_step_k<<<8, 256, 16 * 1536 * sizeof(float), stream>>>(
            HBF, HF32, WHHF, WHHB, GIF, GIB, bhh_f, bhh_b, SUMB, MAXB, t);

    // 5) APPNP: degrees, dinv, z0 = x, then 16 hops (ping-pong, L2-resident)
    deg_k<<<LB(EE), 256, 0, stream>>>(ei, DEG, EE);
    dinv_k<<<LB(NN), 256, 0, stream>>>(DEG, DINV, NN);
    hipMemcpyAsync(Z0, x, (size_t)NN * DD * sizeof(float), hipMemcpyDeviceToDevice, stream);
    for (int hop = 0; hop < 16; hop++) {
        const float* zi = (hop & 1) ? Z1 : Z0;
        float* zo       = (hop & 1) ? Z0 : Z1;
        hop_init_k<<<LB((long)NN * DD), 256, 0, stream>>>(x, zi, DINV, zo, NN * DD);
        hop_edge_k<<<LB((long)EE * (DD / 4)), 256, 0, stream>>>(ei, DINV, zi, zo, EE);
    }
    // after 16 hops (last writes Z0)
    cnt_k<<<LB(NN), 256, 0, stream>>>(batch, CNT, NN);
    pool_k<<<LB((long)NN * DD), 256, 0, stream>>>(batch, Z0, POOL, NN * DD);
    pooldiv_k<<<LB(BB * DD), 256, 0, stream>>>(POOL, CNT, BB * DD);

    // 6) mlp_graph: 768 -> 384 -> 192 -> 14
    linear_k<<<LB(BB * 384), 256, 0, stream>>>(POOL, mg_w0, mg_b0, GH1, BB, 384, 768, 1);
    linear_k<<<LB(BB * 192), 256, 0, stream>>>(GH1, mg_w1, mg_b1, GH2, BB, 192, 384, 1);
    linear_k<<<LB(BB * CC), 256, 0, stream>>>(GH2, mg_w2, mg_b2, GLOG, BB, CC, 192, 0);

    // 7) mlp_seq on (mean + max): 1024 -> 512 -> 256 -> 14
    seqfeat_k<<<LB(BB * 1024), 256, 0, stream>>>(SUMB, MAXB, SFEA, BB * 1024);
    linear_k<<<LB(BB * 512), 256, 0, stream>>>(SFEA, ms_w0, ms_b0, SH1, BB, 512, 1024, 1);
    linear_k<<<LB(BB * 256), 256, 0, stream>>>(SH1, ms_w1, ms_b1, SH2, BB, 256, 512, 1);
    linear_k<<<LB(BB * CC), 256, 0, stream>>>(SH2, ms_w2, ms_b2, SLOG, BB, CC, 256, 0);

    // 8) final: out = g_logits + seq_logits
    add_k<<<LB(BB * CC), 256, 0, stream>>>(GLOG, SLOG, out, BB * CC);
}
